// JesperLLM_74964359184848
// MI455X (gfx1250) — compile-verified, matrix operand order checked by reading, not run on hardware
//
#include <hip/hip_runtime.h>
#include <hip/hip_bf16.h>

// ---------------------------------------------------------------------------
// Model config (from reference)
// ---------------------------------------------------------------------------
#define CFG_V   32000
#define CFG_C   1536
#define CFG_L   2
#define CFG_NH  12
#define CFG_NKV 4
#define CFG_HID 2048
#define CFG_E   8
#define CFG_T   2048
#define CFG_D   128
#define CFG_NREP (CFG_NH / CFG_NKV)

typedef __bf16 bf16;
typedef bf16  v16bf __attribute__((ext_vector_type(16)));
typedef bf16  v8bf  __attribute__((ext_vector_type(8)));
typedef float v8f   __attribute__((ext_vector_type(8)));
typedef unsigned int ui32x4 __attribute__((ext_vector_type(4)));
typedef int   i32x4 __attribute__((ext_vector_type(4)));
typedef int   i32x8 __attribute__((ext_vector_type(8)));

// ---------------------------------------------------------------------------
// WMMA helpers.  CDNA5 bf16 16x16x32 layout (CDNA5 ISA):
//   A frag: lane 0-15 -> M=lane, K = {0..7, 16..23}; lane 16-31 -> M=lane-16,
//           K = {8..15, 24..31}; element 2i/2i+1 in VGPR i.
//   B frag: same per-lane layout with N in place of M (column of B).
//   C/D:    lane owns column N=lane%16; VGPR r holds row r (lanes 0-15) or
//           r+8 (lanes 16-31).
// ---------------------------------------------------------------------------
__device__ __forceinline__ v16bf frag_ld(const bf16* p) {
  v8bf lo = *(const v8bf*)(p);
  v8bf hi = *(const v8bf*)(p + 16);
  v16bf f;
#pragma unroll
  for (int i = 0; i < 8; ++i) { f[i] = lo[i]; f[i + 8] = hi[i]; }
  return f;
}

__device__ __forceinline__ v16bf frag_ld_f32(const float* p) {
  v16bf f;
#pragma unroll
  for (int i = 0; i < 8; ++i) { f[i] = (bf16)p[i]; f[i + 8] = (bf16)p[i + 16]; }
  return f;
}

__device__ __forceinline__ v8f wmma_bf16(v16bf a, v16bf b, v8f c) {
  return __builtin_amdgcn_wmma_f32_16x16x32_bf16(
      /*neg_a=*/false, a, /*neg_b=*/false, b,
      /*c_mod=*/(short)0, c, /*reuse_a=*/false, /*reuse_b=*/false);
}

__device__ __forceinline__ float silu_f(float x) {
  return x / (1.0f + __expf(-x));
}

// ---------------------------------------------------------------------------
// Tensor Data Mover: async-DMA a 2-D fp32 tile [rows x cols] (row stride
// `stride_elems`) from global memory into LDS at byte offset `lds_off`.
// D# layout per CDNA5 ISA; tensor == tile here
// (global_addr already points at the tile start).
// ---------------------------------------------------------------------------
__device__ __forceinline__ void tdm_load_2d_f32(unsigned lds_off,
                                                const float* gptr,
                                                int rows, int cols,
                                                int stride_elems) {
  unsigned long long ga = (unsigned long long)(uintptr_t)gptr;
  ui32x4 g0;
  g0[0] = 1u;                                       // count=1, user desc
  g0[1] = lds_off;                                  // lds_addr (bytes)
  g0[2] = (unsigned)(ga & 0xffffffffu);             // global_addr[31:0]
  g0[3] = (unsigned)((ga >> 32) & 0x01ffffffu)      // global_addr[56:32]
        | (2u << 30);                               // type=2 ("image")
  i32x8 g1;
  g1[0] = (int)(2u << 16);                          // data_size=2 -> 4 bytes
  g1[1] = (int)(((unsigned)cols & 0xffffu) << 16);  // tensor_dim0[15:0]
  g1[2] = (int)((((unsigned)cols >> 16) & 0xffffu)  // tensor_dim0[31:16]
        | (((unsigned)rows & 0xffffu) << 16));      // tensor_dim1[15:0]
  g1[3] = (int)((((unsigned)rows >> 16) & 0xffffu)  // tensor_dim1[31:16]
        | (((unsigned)cols & 0xffffu) << 16));      // tile_dim0
  g1[4] = (int)((unsigned)rows & 0xffffu);          // tile_dim1 (tile_dim2=0)
  g1[5] = stride_elems;                             // tensor_dim0_stride[31:0]
  g1[6] = 0;
  g1[7] = 0;
  i32x4 z4 = {0, 0, 0, 0};
#if __clang_major__ >= 23
  i32x8 z8 = {0, 0, 0, 0, 0, 0, 0, 0};
  __builtin_amdgcn_tensor_load_to_lds(g0, g1, z4, z4, z8, 0);
#else
  __builtin_amdgcn_tensor_load_to_lds(g0, g1, z4, z4, 0);
#endif
}

// ---------------------------------------------------------------------------
// Embedding + positional add
// ---------------------------------------------------------------------------
__global__ void embed_kernel(const int* __restrict__ tokens,
                             const float* __restrict__ emb,
                             const float* __restrict__ pos,
                             float* __restrict__ x) {
  int t = blockIdx.x;
  int tok = tokens[t];
  const float* er = emb + (size_t)tok * CFG_C;
  const float* pr = pos + (size_t)t * CFG_C;
  float* xr = x + (size_t)t * CFG_C;
  for (int c = threadIdx.x; c < CFG_C; c += blockDim.x) xr[c] = er[c] + pr[c];
}

// ---------------------------------------------------------------------------
// RMSNorm: one block (256 thr = 8 waves) per row
// ---------------------------------------------------------------------------
__global__ void rmsnorm_kernel(const float* __restrict__ x,
                               const float* __restrict__ w,
                               float* __restrict__ out) {
  int row = blockIdx.x;
  const float* xr = x + (size_t)row * CFG_C;
  float ss = 0.f;
  for (int c = threadIdx.x; c < CFG_C; c += blockDim.x) {
    float v = xr[c];
    ss += v * v;
  }
#pragma unroll
  for (int off = 16; off >= 1; off >>= 1) ss += __shfl_xor(ss, off, 32);
  __shared__ float red[8];
  int wave = threadIdx.x >> 5, lane = threadIdx.x & 31;
  if (lane == 0) red[wave] = ss;
  __syncthreads();
  if (wave == 0) {
    float t = (lane < 8) ? red[lane] : 0.f;
#pragma unroll
    for (int off = 4; off >= 1; off >>= 1) t += __shfl_xor(t, off, 32);
    if (lane == 0) red[0] = rsqrtf(t / (float)CFG_C + 1e-6f);
  }
  __syncthreads();
  float rn = red[0];
  float* orow = out + (size_t)row * CFG_C;
  for (int c = threadIdx.x; c < CFG_C; c += blockDim.x)
    orow[c] = xr[c] * rn * w[c];
}

// ---------------------------------------------------------------------------
// Router: softmax(E=8) -> top-2 -> renormalized dense weights (wave/token)
// ---------------------------------------------------------------------------
__global__ void gate_topk_kernel(const float* __restrict__ xf,
                                 const float* __restrict__ gate,
                                 float* __restrict__ dense_w) {
  int wave = threadIdx.x >> 5, lane = threadIdx.x & 31;
  int n = blockIdx.x * 8 + wave;
  const float* xr = xf + (size_t)n * CFG_C;
  float acc[CFG_E];
#pragma unroll
  for (int e = 0; e < CFG_E; ++e) acc[e] = 0.f;
  for (int c = lane; c < CFG_C; c += 32) {
    float xv = xr[c];
    const float* gr = gate + (size_t)c * CFG_E;
#pragma unroll
    for (int e = 0; e < CFG_E; ++e) acc[e] += xv * gr[e];
  }
#pragma unroll
  for (int e = 0; e < CFG_E; ++e)
#pragma unroll
    for (int off = 16; off >= 1; off >>= 1)
      acc[e] += __shfl_xor(acc[e], off, 32);
  if (lane == 0) {
    float mx = acc[0];
#pragma unroll
    for (int e = 1; e < CFG_E; ++e) mx = fmaxf(mx, acc[e]);
    float s = 0.f;
#pragma unroll
    for (int e = 0; e < CFG_E; ++e) { acc[e] = __expf(acc[e] - mx); s += acc[e]; }
#pragma unroll
    for (int e = 0; e < CFG_E; ++e) acc[e] /= s;
    int i0 = 0;
#pragma unroll
    for (int e = 1; e < CFG_E; ++e) if (acc[e] > acc[i0]) i0 = e;
    int i1 = (i0 == 0) ? 1 : 0;
#pragma unroll
    for (int e = 0; e < CFG_E; ++e)
      if (e != i0 && acc[e] > acc[i1]) i1 = e;
    float ws = acc[i0] + acc[i1];
    float* dr = dense_w + (size_t)n * CFG_E;
#pragma unroll
    for (int e = 0; e < CFG_E; ++e) dr[e] = 0.f;
    dr[i0] = acc[i0] / ws;
    dr[i1] = acc[i1] / ws;
  }
}

// ---------------------------------------------------------------------------
// MoE routing lists: cnt[e], idx[e*T..] = tokens routed to expert e
// ---------------------------------------------------------------------------
__global__ void moe_zero_cnt(int* __restrict__ cnt) {
  if (threadIdx.x < CFG_E) cnt[threadIdx.x] = 0;
}

__global__ void moe_build_lists(const float* __restrict__ dense_w,
                                int* __restrict__ cnt, int* __restrict__ idx) {
  int n = blockIdx.x * 256 + threadIdx.x;
  const float* dr = dense_w + (size_t)n * CFG_E;
#pragma unroll
  for (int e = 0; e < CFG_E; ++e) {
    if (dr[e] > 0.f) {
      int p = atomicAdd(&cnt[e], 1);
      idx[e * CFG_T + p] = n;
    }
  }
}

// ---------------------------------------------------------------------------
// Dense bf16-WMMA GEMM: C[M,N] = epilogue(A[M,K] * B)
//   TRANS_B=0 : B is [K,N] row-major    TRANS_B=1 : B is [N,K] row-major
// Block tile 256x128, BK=32; 8 waves x (32 rows x 128 cols) = 16 wmma/K-step.
// A tile staged fp32 in LDS via the Tensor Data Mover; B converted to bf16
// on the VALU path.
// ---------------------------------------------------------------------------
enum { EP_STORE = 0, EP_RESID = 1 };

template <bool TRANS_B, int EP>
__global__ __launch_bounds__(256)
void gemm_wmma_kernel(const float* __restrict__ A, const float* __restrict__ B,
                      float* __restrict__ C, const float* __restrict__ R,
                      int M, int N, int Kd, int lda, int ldb, int ldc) {
  __shared__ __align__(16) float Asf[256][32];   // fp32 staging (TDM dest)
  __shared__ __align__(16) bf16  Bs[128][32];    // Bs[n][k]
  (void)M; (void)N;

  const int tid   = threadIdx.x;
  const int wave  = tid >> 5;
  const int lane  = tid & 31;
  const int lm    = lane & 15;
  const int kbsel = (lane < 16) ? 0 : 8;
  const int rAdd  = (lane < 16) ? 0 : 8;
  const int m0 = blockIdx.y * 256;
  const int n0 = blockIdx.x * 128;

  v8f acc[2][8];
#pragma unroll
  for (int s = 0; s < 2; ++s)
#pragma unroll
    for (int i = 0; i < 8; ++i)
#pragma unroll
      for (int r = 0; r < 8; ++r) acc[s][i][r] = 0.f;

  const unsigned lds_a = (unsigned)(uintptr_t)&Asf[0][0];

  for (int k0 = 0; k0 < Kd; k0 += 32) {
    // --- async A tile via TDM (one DMA per block per K-step) ---
    if (wave == 0)
      tdm_load_2d_f32(lds_a, A + (size_t)m0 * lda + k0, 256, 32, lda);
    // --- B tile: fp32 -> bf16 into LDS (all 256 threads) ---
    if (TRANS_B) {
#pragma unroll
      for (int i = 0; i < 16; ++i) {
        int e = tid + i * 256;
        int r = e >> 5, c = e & 31;
        Bs[r][c] = (bf16)B[(size_t)(n0 + r) * ldb + (k0 + c)];
      }
    } else {
#pragma unroll
      for (int i = 0; i < 16; ++i) {
        int e = tid + i * 256;
        int n = e & 127, kk = e >> 7;
        Bs[n][kk] = (bf16)B[(size_t)(k0 + kk) * ldb + (n0 + n)];
      }
    }
    if (k0 + 32 < Kd) {   // prefetch next B tile (global_prefetch_b8)
      if (TRANS_B)
        __builtin_prefetch(&B[(size_t)(n0 + (tid >> 1)) * ldb + (k0 + 32) + (tid & 1) * 16], 0, 3);
      else if (tid < 32)
        __builtin_prefetch(&B[(size_t)(k0 + 32 + tid) * ldb + n0], 0, 3);
    }
    if (wave == 0) __builtin_amdgcn_s_wait_tensorcnt(0);
    __syncthreads();

    v16bf af0 = frag_ld_f32(&Asf[wave * 32 + lm][kbsel]);
    v16bf af1 = frag_ld_f32(&Asf[wave * 32 + 16 + lm][kbsel]);
#pragma unroll
    for (int nn = 0; nn < 8; ++nn) {
      v16bf bfr = frag_ld(&Bs[nn * 16 + lm][kbsel]);
      acc[0][nn] = wmma_bf16(af0, bfr, acc[0][nn]);
      acc[1][nn] = wmma_bf16(af1, bfr, acc[1][nn]);
    }
    __syncthreads();
  }

#pragma unroll
  for (int s = 0; s < 2; ++s)
#pragma unroll
    for (int nn = 0; nn < 8; ++nn) {
      int gn = n0 + nn * 16 + lm;
#pragma unroll
      for (int r = 0; r < 8; ++r) {
        int gm = m0 + wave * 32 + s * 16 + r + rAdd;
        size_t ci = (size_t)gm * ldc + gn;
        float val = acc[s][nn][r];
        if (EP == EP_RESID) val += R[ci];
        C[ci] = val;
      }
    }
}

// ---------------------------------------------------------------------------
// MoE gathered GEMM over the rows routed to one expert.
//   MEP_SILU:    C[i,:]  = silu(A[idx[i],:] @ B)          (compact store)
//   MEP_MUL:     C[i,:] *= A[idx[i],:] @ B                (compact RMW)
//   MEP_SCATTER: C[idx[i],:] += scale[idx[i]]*(A[i,:]@B)  (scatter-add)
// Block tile 128x128; grid.y covers worst case T rows, blocks past cnt exit.
// ---------------------------------------------------------------------------
enum { MEP_SILU = 0, MEP_MUL = 1, MEP_SCATTER = 2 };

template <int MODE>
__global__ __launch_bounds__(256)
void moe_gemm_kernel(const float* __restrict__ A, const float* __restrict__ B,
                     float* __restrict__ C, const float* __restrict__ scale,
                     int sstride, const int* __restrict__ idxE,
                     const int* __restrict__ cntE,
                     int N, int Kd, int lda, int ldb, int ldc) {
  __shared__ __align__(16) bf16 As[128][32];
  __shared__ __align__(16) bf16 Bs[128][32];
  __shared__ int sidx[128];
  (void)N;

  const int cE = *cntE;
  const int m0 = blockIdx.y * 128;
  if (m0 >= cE) return;   // uniform over the block

  const int tid   = threadIdx.x;
  const int wave  = tid >> 5;
  const int lane  = tid & 31;
  const int lm    = lane & 15;
  const int kbsel = (lane < 16) ? 0 : 8;
  const int rAdd  = (lane < 16) ? 0 : 8;
  const int n0 = blockIdx.x * 128;

  if (tid < 128) {
    int r = m0 + tid;
    sidx[tid] = idxE[(r < cE) ? r : (cE - 1)];
  }
  __syncthreads();

  v8f acc[8];
#pragma unroll
  for (int i = 0; i < 8; ++i)
#pragma unroll
    for (int r = 0; r < 8; ++r) acc[i][r] = 0.f;

  for (int k0 = 0; k0 < Kd; k0 += 32) {
#pragma unroll
    for (int i = 0; i < 16; ++i) {
      int e = tid + i * 256;
      int r = e >> 5, c = e & 31;
      int src = (MODE == MEP_SCATTER) ? (m0 + r) : sidx[r];
      As[r][c] = (bf16)A[(size_t)src * lda + (k0 + c)];
    }
#pragma unroll
    for (int i = 0; i < 16; ++i) {
      int e = tid + i * 256;
      int n = e & 127, kk = e >> 7;
      Bs[n][kk] = (bf16)B[(size_t)(k0 + kk) * ldb + (n0 + n)];
    }
    __syncthreads();

    v16bf af = frag_ld(&As[wave * 16 + lm][kbsel]);
#pragma unroll
    for (int nn = 0; nn < 8; ++nn) {
      v16bf bfr = frag_ld(&Bs[nn * 16 + lm][kbsel]);
      acc[nn] = wmma_bf16(af, bfr, acc[nn]);
    }
    __syncthreads();
  }

#pragma unroll
  for (int nn = 0; nn < 8; ++nn) {
    int gn = n0 + nn * 16 + lm;
#pragma unroll
    for (int r = 0; r < 8; ++r) {
      int trow = wave * 16 + r + rAdd;   // row within tile
      if (m0 + trow < cE) {
        float val = acc[nn][r];
        if (MODE == MEP_SILU) {
          C[(size_t)(m0 + trow) * ldc + gn] = silu_f(val);
        } else if (MODE == MEP_MUL) {
          size_t ci = (size_t)(m0 + trow) * ldc + gn;
          C[ci] = C[ci] * val;
        } else {   // MEP_SCATTER
          int g = sidx[trow];
          size_t ci = (size_t)g * ldc + gn;
          C[ci] = C[ci] + scale[(size_t)g * sstride] * val;
        }
      }
    }
  }
}

// ---------------------------------------------------------------------------
// Flash attention (causal, GQA): wave per 16-row query tile, key blocks of
// 32; QK^T and PV on v_wmma_f32_16x16x32_bf16; online softmax with
// intra-16-lane-group shuffles.  grid = (T/64, NH), block = 128.
// ---------------------------------------------------------------------------
__global__ __launch_bounds__(128)
void attn_kernel(const float* __restrict__ q, const float* __restrict__ k,
                 const float* __restrict__ v, float* __restrict__ y) {
  __shared__ __align__(16) bf16 Plds[4][16][32];

  const int wave = threadIdx.x >> 5;
  const int lane = threadIdx.x & 31;
  const int lm    = lane & 15;
  const int kbsel = (lane < 16) ? 0 : 8;
  const int h   = blockIdx.y;
  const int m0  = (blockIdx.x * 4 + wave) * 16;
  const int kvh = h / CFG_NREP;
  const int qstride  = CFG_NH * CFG_D;
  const int kvstride = CFG_NKV * CFG_D;
  const float* qh = q + h * CFG_D;
  const float* kh = k + kvh * CFG_D;
  const float* vh = v + kvh * CFG_D;

  v16bf qa[4];
  {
    const float* qr = qh + (size_t)(m0 + lm) * qstride;
#pragma unroll
    for (int c = 0; c < 4; ++c)
#pragma unroll
      for (int e = 0; e < 8; ++e) {
        qa[c][e]     = (bf16)qr[c * 32 + kbsel + e];
        qa[c][e + 8] = (bf16)qr[c * 32 + kbsel + 16 + e];
      }
  }

  float mrow[8], lrow[8];
  v8f o[8];
#pragma unroll
  for (int r = 0; r < 8; ++r) { mrow[r] = -1e30f; lrow[r] = 0.f; }
#pragma unroll
  for (int j = 0; j < 8; ++j)
#pragma unroll
    for (int r = 0; r < 8; ++r) o[j][r] = 0.f;

  const float sc = 0.08838834764831845f;  // 1/sqrt(128)
  const int kend = m0 + 16;

  for (int kb0 = 0; kb0 < kend; kb0 += 32) {
    v8f s[2];
#pragma unroll
    for (int hh = 0; hh < 2; ++hh) {
      v8f sh;
#pragma unroll
      for (int r = 0; r < 8; ++r) sh[r] = 0.f;
      const float* kr = kh + (size_t)(kb0 + hh * 16 + lm) * kvstride;
#pragma unroll
      for (int c = 0; c < 4; ++c) {
        v16bf kf;
#pragma unroll
        for (int e = 0; e < 8; ++e) {
          kf[e]     = (bf16)kr[c * 32 + kbsel + e];
          kf[e + 8] = (bf16)kr[c * 32 + kbsel + 16 + e];
        }
        sh = wmma_bf16(qa[c], kf, sh);
      }
      s[hh] = sh;
    }
#pragma unroll
    for (int hh = 0; hh < 2; ++hh) {
      int ng = kb0 + hh * 16 + lm;
#pragma unroll
      for (int r = 0; r < 8; ++r) {
        int mg = m0 + r + ((lane < 16) ? 0 : 8);
        float sv = s[hh][r] * sc;
        s[hh][r] = (ng <= mg) ? sv : -1e30f;
      }
    }
#pragma unroll
    for (int r = 0; r < 8; ++r) {
      float mx = fmaxf(s[0][r], s[1][r]);
#pragma unroll
      for (int off = 8; off >= 1; off >>= 1)
        mx = fmaxf(mx, __shfl_xor(mx, off, 32));
      float mn = fmaxf(mrow[r], mx);
      float p0 = __expf(s[0][r] - mn);
      float p1 = __expf(s[1][r] - mn);
      s[0][r] = p0; s[1][r] = p1;
      float ps = p0 + p1;
#pragma unroll
      for (int off = 8; off >= 1; off >>= 1)
        ps += __shfl_xor(ps, off, 32);
      float alpha = __expf(mrow[r] - mn);
      lrow[r] = lrow[r] * alpha + ps;
      mrow[r] = mn;
#pragma unroll
      for (int j = 0; j < 8; ++j) o[j][r] *= alpha;
    }
#pragma unroll
    for (int hh = 0; hh < 2; ++hh)
#pragma unroll
      for (int r = 0; r < 8; ++r)
        Plds[wave][r + ((lane < 16) ? 0 : 8)][hh * 16 + lm] = (bf16)s[hh][r];
    v16bf pf = frag_ld(&Plds[wave][lm][kbsel]);
#pragma unroll
    for (int j = 0; j < 8; ++j) {
      v16bf vf;
      int vcol = j * 16 + lm;
#pragma unroll
      for (int e = 0; e < 8; ++e) {
        vf[e]     = (bf16)vh[(size_t)(kb0 + kbsel + e) * kvstride + vcol];
        vf[e + 8] = (bf16)vh[(size_t)(kb0 + kbsel + 16 + e) * kvstride + vcol];
      }
      o[j] = wmma_bf16(pf, vf, o[j]);
    }
  }

#pragma unroll
  for (int j = 0; j < 8; ++j) {
    int gc = h * CFG_D + j * 16 + lm;
#pragma unroll
    for (int r = 0; r < 8; ++r) {
      int gr = m0 + r + ((lane < 16) ? 0 : 8);
      y[(size_t)gr * qstride + gc] = o[j][r] / lrow[r];
    }
  }
}

// ---------------------------------------------------------------------------
// Host orchestration
// ---------------------------------------------------------------------------
template <bool TB, int EP>
static void launch_gemm(const float* A, const float* B, float* C, const float* R,
                        int M, int N, int Kd, int lda, int ldb, int ldc,
                        hipStream_t s) {
  dim3 grid(N / 128, M / 256);
  gemm_wmma_kernel<TB, EP><<<grid, 256, 0, s>>>(A, B, C, R, M, N, Kd, lda, ldb, ldc);
}

extern "C" void kernel_launch(void* const* d_in, const int* in_sizes, int n_in,
                              void* d_out, int out_size, void* d_ws, size_t ws_size,
                              hipStream_t stream) {
  (void)in_sizes; (void)n_in; (void)out_size; (void)ws_size;
  const int*   tokens  = (const int*)d_in[0];
  const float* emb     = (const float*)d_in[1];
  const float* pos     = (const float*)d_in[2];
  const float* wq      = (const float*)d_in[3];
  const float* wk      = (const float*)d_in[4];
  const float* wv      = (const float*)d_in[5];
  const float* wo      = (const float*)d_in[6];
  const float* attn_nw = (const float*)d_in[7];
  const float* ffn_nw  = (const float*)d_in[8];
  const float* gate    = (const float*)d_in[9];
  const float* ew1     = (const float*)d_in[10];
  const float* ew2     = (const float*)d_in[11];
  const float* ew3     = (const float*)d_in[12];
  const float* fin_nw  = (const float*)d_in[13];
  float* out = (float*)d_out;

  float* ws = (float*)d_ws;
  size_t off = 0;
  float* x  = ws + off; off += (size_t)CFG_T * CFG_C;
  float* hn = ws + off; off += (size_t)CFG_T * CFG_C;
  float* qb = ws + off; off += (size_t)CFG_T * CFG_NH  * CFG_D;
  float* kb = ws + off; off += (size_t)CFG_T * CFG_NKV * CFG_D;
  float* vb = ws + off; off += (size_t)CFG_T * CFG_NKV * CFG_D;
  float* yb = ws + off; off += (size_t)CFG_T * CFG_C;
  float* gw = ws + off; off += (size_t)CFG_T * CFG_E;
  float* hb = ws + off; off += (size_t)CFG_T * CFG_HID;
  int*   cnt = (int*)(ws + off); off += CFG_E;
  int*   idx = (int*)(ws + off); off += (size_t)CFG_E * CFG_T;

  embed_kernel<<<CFG_T, 256, 0, stream>>>(tokens, emb, pos, x);

  const int CQ = CFG_NH * CFG_D;    // 1536
  const int CK = CFG_NKV * CFG_D;   // 512

  for (int l = 0; l < CFG_L; ++l) {
    // --- attention block ---
    rmsnorm_kernel<<<CFG_T, 256, 0, stream>>>(x, attn_nw + (size_t)l * CFG_C, hn);
    launch_gemm<false, EP_STORE>(hn, wq + (size_t)l * CFG_C * CQ, qb, nullptr,
                                 CFG_T, CQ, CFG_C, CFG_C, CQ, CQ, stream);
    launch_gemm<false, EP_STORE>(hn, wk + (size_t)l * CFG_C * CK, kb, nullptr,
                                 CFG_T, CK, CFG_C, CFG_C, CK, CK, stream);
    launch_gemm<false, EP_STORE>(hn, wv + (size_t)l * CFG_C * CK, vb, nullptr,
                                 CFG_T, CK, CFG_C, CFG_C, CK, CK, stream);
    attn_kernel<<<dim3(CFG_T / 64, CFG_NH), 128, 0, stream>>>(qb, kb, vb, yb);
    launch_gemm<false, EP_RESID>(yb, wo + (size_t)l * CFG_C * CFG_C, x, x,
                                 CFG_T, CFG_C, CFG_C, CFG_C, CFG_C, CFG_C, stream);

    // --- MoE block: route, then compute only the selected rows ---
    rmsnorm_kernel<<<CFG_T, 256, 0, stream>>>(x, ffn_nw + (size_t)l * CFG_C, hn);
    gate_topk_kernel<<<CFG_T / 8, 256, 0, stream>>>(hn, gate + (size_t)l * CFG_C * CFG_E, gw);
    moe_zero_cnt<<<1, 32, 0, stream>>>(cnt);
    moe_build_lists<<<CFG_T / 256, 256, 0, stream>>>(gw, cnt, idx);
    for (int e = 0; e < CFG_E; ++e) {
      const float* w1 = ew1 + ((size_t)l * CFG_E + e) * CFG_C * CFG_HID;
      const float* w3 = ew3 + ((size_t)l * CFG_E + e) * CFG_C * CFG_HID;
      const float* w2 = ew2 + ((size_t)l * CFG_E + e) * CFG_HID * CFG_C;
      const int* idxE = idx + (size_t)e * CFG_T;
      const int* cntE = cnt + e;
      dim3 g1g(CFG_HID / 128, CFG_T / 128);
      moe_gemm_kernel<MEP_SILU><<<g1g, 256, 0, stream>>>(
          hn, w1, hb, nullptr, 0, idxE, cntE, CFG_HID, CFG_C, CFG_C, CFG_HID, CFG_HID);
      moe_gemm_kernel<MEP_MUL><<<g1g, 256, 0, stream>>>(
          hn, w3, hb, nullptr, 0, idxE, cntE, CFG_HID, CFG_C, CFG_C, CFG_HID, CFG_HID);
      dim3 g2g(CFG_C / 128, CFG_T / 128);
      moe_gemm_kernel<MEP_SCATTER><<<g2g, 256, 0, stream>>>(
          hb, w2, x, gw + e, CFG_E, idxE, cntE, CFG_C, CFG_HID, CFG_HID, CFG_C, CFG_C);
    }
  }

  rmsnorm_kernel<<<CFG_T, 256, 0, stream>>>(x, fin_nw, hn);
  launch_gemm<true, EP_STORE>(hn, emb, out, nullptr,
                              CFG_T, CFG_V, CFG_C, CFG_C, CFG_C, CFG_V, stream);
}
